// M_36283883716705
// MI455X (gfx1250) — matrix operand order repaired and checked by reading, not compiled, not run on hardware
//
#include <hip/hip_runtime.h>

#define SEQ   2048
#define NH    32
#define NKV   8
#define HD    128
#define GROUP 4
#define SCALE 0.08838834764831845f
#define NEG_INF (-__builtin_huge_valf())

typedef __attribute__((ext_vector_type(16))) __bf16 v16bf;
typedef __attribute__((ext_vector_type(8)))  __bf16 v8bf;
typedef __attribute__((ext_vector_type(8)))  float  v8f;
typedef __attribute__((ext_vector_type(4)))  float  v4f;
typedef __attribute__((ext_vector_type(4)))  unsigned int v4u;
typedef __attribute__((ext_vector_type(4)))  int v4i;
typedef __attribute__((ext_vector_type(8)))  int v8i;

union BF16x16 { v16bf v; v8bf h[2]; unsigned short us[16]; };

__device__ __forceinline__ unsigned short f2bf(float x) {
  return __builtin_bit_cast(unsigned short, (__bf16)x);   // native cvt
}

// max across the 16-lane group (wave32) via ds_swizzle immediates
__device__ __forceinline__ float swz_max16(float x) {
  int i;
  i = __builtin_amdgcn_ds_swizzle(__builtin_bit_cast(int, x), 0x041f); // xor 1
  x = fmaxf(x, __builtin_bit_cast(float, i));
  i = __builtin_amdgcn_ds_swizzle(__builtin_bit_cast(int, x), 0x081f); // xor 2
  x = fmaxf(x, __builtin_bit_cast(float, i));
  i = __builtin_amdgcn_ds_swizzle(__builtin_bit_cast(int, x), 0x101f); // xor 4
  x = fmaxf(x, __builtin_bit_cast(float, i));
  i = __builtin_amdgcn_ds_swizzle(__builtin_bit_cast(int, x), 0x201f); // xor 8
  x = fmaxf(x, __builtin_bit_cast(float, i));
  return x;
}

#define HAVE_TDM __has_builtin(__builtin_amdgcn_tensor_load_to_lds)

// 2D TDM tile load: (tile0 x tile1) bf16 tile of tensor (td0 x td1, row stride
// stride0 elements) -> contiguous LDS at lds_addr. D# per CDNA5 ISA.
__device__ __forceinline__ void tdm_load_2d(unsigned lds_addr, const unsigned short* g,
                                            unsigned td0, unsigned td1,
                                            unsigned tile0, unsigned tile1,
                                            unsigned stride0) {
#if HAVE_TDM
  unsigned long long ga = (unsigned long long)(uintptr_t)g;
  v4u g0 = { 1u,                                  // count=1 (valid descriptor)
             lds_addr,                            // lds_addr [63:32]
             (unsigned)ga,                        // global_addr lo
             (unsigned)((ga >> 32) & 0x01ffffffu) | (2u << 30) }; // ga hi | type=2
  v8i g1 = { (int)0x00010000u,                              // data_size=2B
             (int)((td0 & 0xffffu) << 16),                  // tensor_dim0 lo
             (int)((td0 >> 16) | ((td1 & 0xffffu) << 16)),  // td0 hi | td1 lo
             (int)((td1 >> 16) | (tile0 << 16)),            // td1 hi | tile_dim0
             (int)tile1,                                    // tile_dim1 (tile_dim2=0)
             (int)stride0,                                  // dim0 stride lo
             0, 0 };
  v4i z4 = {0, 0, 0, 0};
#if __clang_major__ >= 23
  v8i z8 = {0, 0, 0, 0, 0, 0, 0, 0};
  __builtin_amdgcn_tensor_load_to_lds(g0, g1, z4, z4, z8, 0);
#else
  __builtin_amdgcn_tensor_load_to_lds(g0, g1, z4, z4, 0);
#endif
#else
  (void)lds_addr; (void)g; (void)td0; (void)td1; (void)tile0; (void)tile1; (void)stride0;
#endif
}

// ---- pre-pass: K fp32 -> bf16 row-major -------------------------------------
__global__ void convert_k_bf16(const float* __restrict__ k, unsigned short* __restrict__ kbf) {
  size_t i = (size_t)blockIdx.x * 256 + threadIdx.x;
  kbf[i] = f2bf(k[i]);
}

// ---- pre-pass: V fp32 [S][1024] -> bf16 transposed [1024][S] (tiled) --------
__global__ void transpose_v_bf16(const float* __restrict__ v, unsigned short* __restrict__ vtg) {
  __shared__ float tile[32][33];
  const int c0 = blockIdx.x * 32, s0 = blockIdx.y * 32;
  const int tx = threadIdx.x, ty = threadIdx.y;   // block (32, 8)
  #pragma unroll
  for (int j = 0; j < 32; j += 8)
    tile[ty + j][tx] = v[(size_t)(s0 + ty + j) * (NKV * HD) + c0 + tx];
  __syncthreads();
  #pragma unroll
  for (int j = 0; j < 32; j += 8)
    vtg[(size_t)(c0 + ty + j) * SEQ + s0 + tx] = f2bf(tile[tx][ty + j]);
}

// ---- main flash-attention kernel --------------------------------------------
__global__ __launch_bounds__(128, 1)
void fa_gqa_bf16_kernel(const float* __restrict__ q,
                        const unsigned short* __restrict__ kbf,   // bf16 [S][1024]
                        const unsigned short* __restrict__ vtg,   // bf16 [1024][S]
                        float* __restrict__ out) {
  __shared__ __attribute__((aligned(32))) unsigned short Kb[2][32 * 128]; // K tile row-major
  __shared__ __attribute__((aligned(32))) unsigned short Vt[2][128 * 32]; // V tile transposed
  __shared__ __attribute__((aligned(32))) unsigned short Pb[4 * 16 * 32]; // per-wave P bounce

  const int qblk = blockIdx.x;
  const int head = blockIdx.y;
  const int kvh  = head / GROUP;
  const int tid  = threadIdx.x;
  const int wave = tid >> 5;
  const int lane = tid & 31;
  const int half = lane >> 4;
  const int l16  = lane & 15;

  const int  q0w    = qblk * 64 + wave * 16;
  const int  kend   = qblk * 64 + 64;
  const int  qmax_s = __builtin_amdgcn_readfirstlane(q0w) + 15; // SGPR: scalar branch
  const bool isw0   = __builtin_amdgcn_readfirstlane(wave) == 0;

  // Q tile (16x128) as 4 bf16 A-fragments (vectorized f32 loads, native cvt)
  v16bf qf[4];
  {
    const int qrow = q0w + l16;
    const float* qp = q + (size_t)qrow * (NH * HD) + head * HD;
    #pragma unroll
    for (int c = 0; c < 4; ++c) {
      BF16x16 t;
      const v4f x0 = *(const v4f*)(qp + c * 32 + half * 16);
      const v4f x1 = *(const v4f*)(qp + c * 32 + half * 16 + 4);
      const v4f x2 = *(const v4f*)(qp + c * 32 + half * 16 + 8);
      const v4f x3 = *(const v4f*)(qp + c * 32 + half * 16 + 12);
      #pragma unroll
      for (int e = 0; e < 4; ++e) {
        t.us[e]      = f2bf(x0[e]);
        t.us[4 + e]  = f2bf(x1[e]);
        t.us[8 + e]  = f2bf(x2[e]);
        t.us[12 + e] = f2bf(x3[e]);
      }
      qf[c] = t.v;
    }
  }

  BF16x16 ones;                 // B = all-ones: row-sum via WMMA
  #pragma unroll
  for (int e = 0; e < 16; ++e) ones.us[e] = 0x3f80;

  v8f acc[8];
  #pragma unroll
  for (int t = 0; t < 8; ++t) acc[t] = (v8f){};
  float m_i[8], l_i[8];
  #pragma unroll
  for (int r = 0; r < 8; ++r) { m_i[r] = NEG_INF; l_i[r] = 0.f; }

  unsigned short* Pw = Pb + wave * (16 * 32);

  auto load_tile = [&](int pp, int kbb) {
#if HAVE_TDM
    if (isw0) {
      tdm_load_2d((unsigned)(uintptr_t)&Kb[pp][0],
                  kbf + (size_t)kbb * (NKV * HD) + kvh * HD,
                  NKV * HD, SEQ, HD, 32, NKV * HD);
      tdm_load_2d((unsigned)(uintptr_t)&Vt[pp][0],
                  vtg + (size_t)(kvh * HD) * SEQ + kbb,
                  SEQ, NKV * HD, 32, HD, SEQ);
    }
#else
    #pragma unroll 4
    for (int i = 0; i < 32; ++i) {
      Kb[pp][i * 128 + tid] = kbf[(size_t)(kbb + i) * (NKV * HD) + kvh * HD + tid];
      Vt[pp][tid * 32 + i]  = vtg[(size_t)(kvh * HD + tid) * SEQ + kbb + i];
    }
#endif
  };

  load_tile(0, 0);              // prologue: block 0 -> buffer 0
  int p = 0;

  for (int kb = 0; kb < kend; kb += 32) {
    __syncthreads();            // all reads of buf(1-p) from prev iter done
    const bool has_next = (kb + 32) < kend;
    if (has_next) load_tile(p ^ 1, kb + 32);
#if HAVE_TDM
    if (isw0) {
      if (has_next) __builtin_amdgcn_s_wait_tensorcnt(2); // pair(kb) done
      else          __builtin_amdgcn_s_wait_tensorcnt(0);
    }
#endif
    __syncthreads();            // buf(p) visible to all waves

    if (kb <= qmax_s) {         // scalar branch: EXEC stays all-ones
      const unsigned short* Kp = &Kb[p][0];
      const unsigned short* Vp = &Vt[p][0];
      const int rowK = l16 * 128 + half * 16;

      // preload ALL K B-fragments into distinct regs -> incremental dscnt waits
      v16bf kf0 = *(const v16bf*)&Kp[rowK + 0];
      v16bf kf1 = *(const v16bf*)&Kp[rowK + 32];
      v16bf kf2 = *(const v16bf*)&Kp[rowK + 64];
      v16bf kf3 = *(const v16bf*)&Kp[rowK + 96];
      v16bf kg0 = *(const v16bf*)&Kp[2048 + rowK + 0];
      v16bf kg1 = *(const v16bf*)&Kp[2048 + rowK + 32];
      v16bf kg2 = *(const v16bf*)&Kp[2048 + rowK + 64];
      v16bf kg3 = *(const v16bf*)&Kp[2048 + rowK + 96];

      v8f s0 = (v8f){}, s1 = (v8f){};
      s0 = __builtin_amdgcn_wmma_f32_16x16x32_bf16(false, qf[0], false, kf0, (short)0, s0, false, false);
      s1 = __builtin_amdgcn_wmma_f32_16x16x32_bf16(false, qf[0], false, kg0, (short)0, s1, false, false);
      s0 = __builtin_amdgcn_wmma_f32_16x16x32_bf16(false, qf[1], false, kf1, (short)0, s0, false, false);
      s1 = __builtin_amdgcn_wmma_f32_16x16x32_bf16(false, qf[1], false, kg1, (short)0, s1, false, false);
      s0 = __builtin_amdgcn_wmma_f32_16x16x32_bf16(false, qf[2], false, kf2, (short)0, s0, false, false);
      s1 = __builtin_amdgcn_wmma_f32_16x16x32_bf16(false, qf[2], false, kg2, (short)0, s1, false, false);
      s0 = __builtin_amdgcn_wmma_f32_16x16x32_bf16(false, qf[3], false, kf3, (short)0, s0, false, false);
      s1 = __builtin_amdgcn_wmma_f32_16x16x32_bf16(false, qf[3], false, kg3, (short)0, s1, false, false);

      // preload V B-fragments NOW; ds latency hides under softmax VALU below
      v16bf vf[8];
      #pragma unroll
      for (int t = 0; t < 8; ++t)
        vf[t] = *(const v16bf*)&Vp[(t * 16 + l16) * 32 + half * 16];

      // mask + online max + exp; stage P (bf16) into LDS
      const int key0 = kb + l16;
      const int key1 = kb + 16 + l16;
      float fscale[8];
      #pragma unroll
      for (int r = 0; r < 8; ++r) {
        const int qrow = q0w + r + half * 8;
        const float a = (key0 <= qrow) ? s0[r] * SCALE : NEG_INF;
        const float b = (key1 <= qrow) ? s1[r] * SCALE : NEG_INF;
        const float m_new = fmaxf(m_i[r], swz_max16(fmaxf(a, b)));
        const float e0 = __expf(a - m_new);
        const float e1 = __expf(b - m_new);
        fscale[r] = __expf(m_i[r] - m_new);
        m_i[r] = m_new;
        const int M = r + half * 8;
        Pw[M * 32 + l16]      = f2bf(e0);
        Pw[M * 32 + 16 + l16] = f2bf(e1);
      }

      // rescale O while P bounce is in flight
      #pragma unroll
      for (int t = 0; t < 8; ++t)
        #pragma unroll
        for (int r = 0; r < 8; ++r) acc[t][r] *= fscale[r];

      // P as A-fragment (wave-local LDS bounce, in-order per wave)
      BF16x16 pa;
      pa.h[0] = *(const v8bf*)&Pw[l16 * 32 + half * 16];
      pa.h[1] = *(const v8bf*)&Pw[l16 * 32 + half * 16 + 8];

      // row-sum(P) via one WMMA: P(16x32) * ones(32x16) -> sums in every column
      v8f rs = __builtin_amdgcn_wmma_f32_16x16x32_bf16(false, pa.v, false, ones.v,
                                                       (short)0, (v8f){}, false, false);
      #pragma unroll
      for (int r = 0; r < 8; ++r) l_i[r] = l_i[r] * fscale[r] + rs[r];

      // O += P * V
      #pragma unroll
      for (int t = 0; t < 8; ++t)
        acc[t] = __builtin_amdgcn_wmma_f32_16x16x32_bf16(false, pa.v, false, vf[t],
                                                         (short)0, acc[t], false, false);
    }
    p ^= 1;
  }

  // epilogue: normalize, store fp32 (coalesced across the 16-lane group)
  #pragma unroll
  for (int r = 0; r < 8; ++r) {
    const float invl = 1.f / l_i[r];
    const int row = q0w + r + half * 8;
    float* op = out + (size_t)row * (NH * HD) + head * HD + l16;
    #pragma unroll
    for (int t = 0; t < 8; ++t) op[t * 16] = acc[t][r] * invl;
  }
}

extern "C" void kernel_launch(void* const* d_in, const int* in_sizes, int n_in,
                              void* d_out, int out_size, void* d_ws, size_t ws_size,
                              hipStream_t stream) {
  (void)in_sizes; (void)n_in; (void)out_size; (void)ws_size;
  const float* q = (const float*)d_in[0];
  const float* k = (const float*)d_in[1];
  const float* v = (const float*)d_in[2];
  float* out = (float*)d_out;

  unsigned short* kbf = (unsigned short*)d_ws;                    // 4 MB
  unsigned short* vtg = kbf + (size_t)SEQ * NKV * HD;             // 4 MB

  convert_k_bf16<<<(SEQ * NKV * HD) / 256, 256, 0, stream>>>(k, kbf);
  transpose_v_bf16<<<dim3((NKV * HD) / 32, SEQ / 32), dim3(32, 8), 0, stream>>>(v, vtg);
  fa_gqa_bf16_kernel<<<dim3(SEQ / 64, NH), dim3(128), 0, stream>>>(q, kbf, vtg, out);
}
